// Layer_Attention_43602507989211
// MI455X (gfx1250) — matrix operand order repaired and checked by reading, not compiled, not run on hardware
//
#include <hip/hip_runtime.h>
#include <hip/hip_bf16.h>
#include <stdint.h>

typedef __attribute__((ext_vector_type(16))) __bf16 bf16x16;
typedef __attribute__((ext_vector_type(8)))  __bf16 bf16x8;
typedef __attribute__((ext_vector_type(8)))  float  f32x8;

union frag_t { bf16x16 v; bf16x8 h[2]; };

#define DIM   64
#define NH    8
#define BATCH 4
#define HIMG  256
#define WIMG  256
#define NPIX  (HIMG*WIMG)     // 65536
#define WSUB  (WIMG/NH)       // 32

// ---- workspace layout (bytes) ----
#define OFF_WQ   0u
#define OFF_WK   8192u
#define OFF_WV   16384u
#define OFF_WP   24576u
#define OFF_BQ   32768u
#define OFF_BK   33024u
#define OFF_BV   33280u
#define OFF_Q    65536u
#define QK_BYTES (BATCH*DIM*HIMG*WSUB*2u)          // 4 MiB
#define OFF_K    (OFF_Q + QK_BYTES)
#define OFF_V0   (OFF_K + QK_BYTES)                // also reused as "out" (pre-proj)
#define PLANE_BYTES ((size_t)BATCH*DIM*NPIX*2u)    // 32 MiB
#define OFF_V    (OFF_V0 + PLANE_BYTES)

#if defined(__AMDGCN__) && __has_builtin(__builtin_amdgcn_global_load_async_to_lds_b128)
#define HAVE_ASYNC_LDS 1
#else
#define HAVE_ASYNC_LDS 0
#endif

__device__ __forceinline__ f32x8 wmma_bf16(bf16x16 a, bf16x16 b, f32x8 c) {
    return __builtin_amdgcn_wmma_f32_16x16x32_bf16(false, a, false, b, (short)0, c, false, false);
}

// ---------------- Kernel 1: fold weights ----------------
__global__ void fold_weights(const float* __restrict__ w_qkv, const float* __restrict__ b_qkv,
                             const float* __restrict__ w_q,   const float* __restrict__ b_q,
                             const float* __restrict__ w_k,   const float* __restrict__ b_k,
                             const float* __restrict__ w_proj, char* ws) {
    __bf16* Wq = (__bf16*)(ws + OFF_WQ);
    __bf16* Wk = (__bf16*)(ws + OFF_WK);
    __bf16* Wv = (__bf16*)(ws + OFF_WV);
    __bf16* Wp = (__bf16*)(ws + OFF_WP);
    float*  bq = (float*)(ws + OFF_BQ);
    float*  bk = (float*)(ws + OFF_BK);
    float*  bv = (float*)(ws + OFF_BV);
    int t = threadIdx.x;
    for (int idx = t; idx < DIM*DIM; idx += 256) {
        int o = idx >> 6, c = idx & 63;
        float aq = 0.f, ak = 0.f;
        for (int m = 0; m < DIM; ++m) {
            aq += w_q[o*DIM + m] * w_qkv[m*DIM + c];
            ak += w_k[o*DIM + m] * w_qkv[(DIM + m)*DIM + c];
        }
        Wq[idx] = (__bf16)aq;
        Wk[idx] = (__bf16)ak;
        Wv[idx] = (__bf16)w_qkv[(2*DIM + o)*DIM + c];
        Wp[idx] = (__bf16)w_proj[idx];
    }
    if (t < DIM) {
        float aq = b_q[t], ak = b_k[t];
        for (int m = 0; m < DIM; ++m) {
            aq += w_q[t*DIM + m] * b_qkv[m];
            ak += w_k[t*DIM + m] * b_qkv[DIM + m];
        }
        bq[t] = aq; bk[t] = ak; bv[t] = b_qkv[2*DIM + t];
    }
}

// ---------------- Kernel 2: q/k at strided columns ----------------
// q stored [b][c][i][h] (row-major along h); k stored TRANSPOSED [b][c][h][j]
// so both are row-major A-matrices for their WMMA consumers. temperature folded into q.
__global__ void qk_kernel(const float* __restrict__ x, const float* __restrict__ temperature,
                          char* ws) {
    const __bf16* Wq = (const __bf16*)(ws + OFF_WQ);
    const __bf16* Wk = (const __bf16*)(ws + OFF_WK);
    const float*  bq = (const float*)(ws + OFF_BQ);
    const float*  bk = (const float*)(ws + OFF_BK);
    __bf16* qo = (__bf16*)(ws + OFF_Q);
    __bf16* ko = (__bf16*)(ws + OFF_K);

    __shared__ float xs[8][DIM];
    int tid = threadIdx.x, wave = tid >> 5, lane = tid & 31;
    int p = blockIdx.x * 8 + wave;           // B*256*32 = 32768 pixels
    int b = p >> 13; int rem = p & 8191; int i = rem >> 5; int h = rem & 31;

    const float* xp = x + (size_t)(b*DIM)*NPIX + i*WIMG + h*NH;
    xs[wave][lane]      = xp[(size_t)lane*NPIX];
    xs[wave][lane + 32] = xp[(size_t)(lane + 32)*NPIX];
    __syncthreads();

    float aq0 = bq[lane], aq1 = bq[lane + 32];
    float ak0 = bk[lane], ak1 = bk[lane + 32];
    #pragma unroll 8
    for (int m = 0; m < DIM; ++m) {
        float xv = xs[wave][m];
        aq0 += (float)Wq[lane*DIM + m]        * xv;
        aq1 += (float)Wq[(lane + 32)*DIM + m] * xv;
        ak0 += (float)Wk[lane*DIM + m]        * xv;
        ak1 += (float)Wk[(lane + 32)*DIM + m] * xv;
    }
    aq0 *= temperature[lane]; aq1 *= temperature[lane + 32];

    size_t pl = HIMG * WSUB;                  // 8192 per (b,c) plane
    qo[(size_t)(b*DIM + lane)*pl      + (size_t)i*WSUB + h] = (__bf16)aq0;
    qo[(size_t)(b*DIM + lane + 32)*pl + (size_t)i*WSUB + h] = (__bf16)aq1;
    ko[(size_t)(b*DIM + lane)*pl      + (size_t)h*HIMG + i] = (__bf16)ak0;  // transposed
    ko[(size_t)(b*DIM + lane + 32)*pl + (size_t)h*HIMG + i] = (__bf16)ak1;  // transposed
}

// ---------------- Kernel 3: v0 = Wv @ x + bv  (bf16 WMMA) ----------------
__global__ void v0_wmma(const float* __restrict__ x, char* ws) {
    __shared__ __attribute__((aligned(32))) __bf16 sW[DIM*DIM];
    const __bf16* Wv = (const __bf16*)(ws + OFF_WV);
    const float*  bv = (const float*)(ws + OFF_BV);
    __bf16* v0 = (__bf16*)(ws + OFF_V0);

    int tid = threadIdx.x, wave = tid >> 5, lane = tid & 31;
    int hi = lane >> 4, lm = lane & 15;
    for (int idx = tid; idx < DIM*DIM; idx += 256) sW[idx] = Wv[idx];
    __syncthreads();

    int b    = blockIdx.x >> 11;             // 2048 blocks per image
    int pix0 = (blockIdx.x & 2047) * 32;
    int mt = wave & 3, nt = wave >> 2;       // c_out tile, pixel half

    f32x8 acc;
    #pragma unroll
    for (int v = 0; v < 8; ++v) acc[v] = bv[mt*16 + v + 8*hi];

    #pragma unroll
    for (int kk = 0; kk < DIM; kk += 32) {
        frag_t a, bf;
        a.h[0] = *(const bf16x8*)&sW[(mt*16 + lm)*DIM + kk + hi*8];
        a.h[1] = *(const bf16x8*)&sW[(mt*16 + lm)*DIM + kk + 16 + hi*8];
        const float* xp = x + (size_t)(b*DIM + kk)*NPIX + pix0 + nt*16 + lm;
        #pragma unroll
        for (int e = 0; e < 16; ++e) bf.v[e] = (__bf16)xp[(size_t)(hi*8 + e + (e & 8))*NPIX];
        acc = wmma_bf16(a.v, bf.v, acc);
    }
    #pragma unroll
    for (int v = 0; v < 8; ++v) {
        int co = mt*16 + v + 8*hi;
        v0[(size_t)(b*DIM + co)*NPIX + pix0 + nt*16 + lm] = (__bf16)acc[v];
    }
}

// ---------------- Kernel 4: 3x3 depthwise on v0 -> v ----------------
__global__ void dw3x3(const float* __restrict__ w_dwv, const float* __restrict__ b_dwv, char* ws) {
    const __bf16* v0 = (const __bf16*)(ws + OFF_V0);
    __bf16* v = (__bf16*)(ws + OFF_V);
    size_t idx = (size_t)blockIdx.x*256 + threadIdx.x;   // < 16,777,216
    int w    = (int)(idx & 255);
    int hrow = (int)((idx >> 8) & 255);
    int c    = (int)((idx >> 16) & 63);
    const __bf16* plane = v0 + (idx & ~(size_t)0xFFFF);
    float acc = b_dwv[c];
    #pragma unroll
    for (int dy = -1; dy <= 1; ++dy)
        #pragma unroll
        for (int dx = -1; dx <= 1; ++dx) {
            int yy = hrow + dy, xx = w + dx;
            if (yy >= 0 && yy < HIMG && xx >= 0 && xx < WIMG)
                acc += w_dwv[c*9 + (dy + 1)*3 + (dx + 1)] * (float)plane[yy*WIMG + xx];
        }
    v[idx] = (__bf16)acc;
}

// ---------------- Kernel 5: per-(b,c)  M = k^T v ; out = q M ----------------
__global__ void attn_wmma(char* ws) {
    __shared__ __attribute__((aligned(32))) __bf16 sQ[HIMG*WSUB];   // q[i][h] 256x32
    __shared__ __attribute__((aligned(32))) __bf16 sK[WSUB*HIMG];   // k[h][j] 32x256 (pre-transposed)
    __shared__ __attribute__((aligned(32))) __bf16 sM[WSUB*WIMG];   // M[h][w] 32x256

    int tid = threadIdx.x, wave = tid >> 5, lane = tid & 31;
    int hi = lane >> 4, lm = lane & 15;
    int bc = blockIdx.x;               // (b,c) pair, 0..255

    const __bf16* qp = (const __bf16*)(ws + OFF_Q) + (size_t)bc * (HIMG*WSUB);
    const __bf16* kp = (const __bf16*)(ws + OFF_K) + (size_t)bc * (HIMG*WSUB);
    const __bf16* vp = (const __bf16*)(ws + OFF_V) + (size_t)bc * NPIX;
    __bf16*       op = (__bf16*)(ws + OFF_V0)      + (size_t)bc * NPIX;   // reuse v0 region

#if HAVE_ASYNC_LDS
    {   // CDNA5 async global->LDS staging of q and k planes (ASYNCcnt path)
        typedef int v4i __attribute__((vector_size(4 * sizeof(int))));
        typedef __attribute__((address_space(1))) v4i gv4i;
        typedef __attribute__((address_space(3))) v4i lv4i;
        gv4i* gq = (gv4i*)(void*)qp;
        gv4i* gk = (gv4i*)(void*)kp;
        lv4i* lq = (lv4i*)(void*)&sQ[0];
        lv4i* lk = (lv4i*)(void*)&sK[0];
        #pragma unroll
        for (int c = 0; c < 4; ++c) {   // 4 passes x 256 threads x 16B = 16 KB per plane
            __builtin_amdgcn_global_load_async_to_lds_b128(gq + c*256 + tid, lq + c*256 + tid, 0, 0);
            __builtin_amdgcn_global_load_async_to_lds_b128(gk + c*256 + tid, lk + c*256 + tid, 0, 0);
        }
#if __has_builtin(__builtin_amdgcn_s_wait_asynccnt)
        __builtin_amdgcn_s_wait_asynccnt(0);
#else
        asm volatile("s_wait_asynccnt 0x0" ::: "memory");
#endif
    }
#else
    for (int idx = tid*8; idx < HIMG*WSUB; idx += 256*8) {
        *(bf16x8*)&sQ[idx] = *(const bf16x8*)&qp[idx];
        *(bf16x8*)&sK[idx] = *(const bf16x8*)&kp[idx];
    }
#endif
    __syncthreads();

    // Phase 1: M[h][w] = sum_j k^T[h][j] * v[j][w].  B (v tile) shared by mt=0/1.
    #pragma unroll
    for (int nt2 = 0; nt2 < 2; ++nt2) {
        int nt = wave*2 + nt2;                         // w tile, 0..15 across block
        f32x8 acc0 = {}, acc1 = {};
        for (int jb = 0; jb < HIMG; jb += 32) {
            frag_t b, a0, a1;
            #pragma unroll
            for (int e = 0; e < 16; ++e) b.v[e] = vp[(size_t)(jb + hi*8 + e + (e & 8))*WIMG + nt*16 + lm];
            a0.h[0] = *(const bf16x8*)&sK[(size_t)lm*HIMG        + jb + hi*8];
            a0.h[1] = *(const bf16x8*)&sK[(size_t)lm*HIMG        + jb + 16 + hi*8];
            a1.h[0] = *(const bf16x8*)&sK[(size_t)(16 + lm)*HIMG + jb + hi*8];
            a1.h[1] = *(const bf16x8*)&sK[(size_t)(16 + lm)*HIMG + jb + 16 + hi*8];
            acc0 = wmma_bf16(a0.v, b.v, acc0);
            acc1 = wmma_bf16(a1.v, b.v, acc1);
        }
        #pragma unroll
        for (int v = 0; v < 8; ++v) {
            sM[(size_t)(v + 8*hi)*WIMG      + nt*16 + lm] = (__bf16)acc0[v];
            sM[(size_t)(16 + v + 8*hi)*WIMG + nt*16 + lm] = (__bf16)acc1[v];
        }
    }
    __syncthreads();

    // Phase 2: out[i][w] = sum_h q[i][h] * M[h][w]  (K=32, one WMMA per tile; A hoisted)
    #pragma unroll
    for (int it2 = 0; it2 < 2; ++it2) {
        int it = wave*2 + it2;                         // i tile, 0..15 across block
        frag_t a;
        a.h[0] = *(const bf16x8*)&sQ[(size_t)(it*16 + lm)*WSUB + hi*8];
        a.h[1] = *(const bf16x8*)&sQ[(size_t)(it*16 + lm)*WSUB + 16 + hi*8];
        for (int wt = 0; wt < 16; ++wt) {
            frag_t b;
            #pragma unroll
            for (int e = 0; e < 16; ++e) b.v[e] = sM[(size_t)(hi*8 + e + (e & 8))*WIMG + wt*16 + lm];
            f32x8 acc = {};
            acc = wmma_bf16(a.v, b.v, acc);
            #pragma unroll
            for (int v = 0; v < 8; ++v)
                op[(size_t)(it*16 + v + 8*hi)*WIMG + wt*16 + lm] = (__bf16)acc[v];
        }
    }
}

// ---------------- Kernel 6: final projection (bf16 WMMA), writes f32 d_out ----------------
__global__ void proj_wmma(const float* __restrict__ b_proj, float* __restrict__ out, char* ws) {
    __shared__ __attribute__((aligned(32))) __bf16 sW[DIM*DIM];
    const __bf16* Wp = (const __bf16*)(ws + OFF_WP);
    const __bf16* in = (const __bf16*)(ws + OFF_V0);   // pre-proj result

    int tid = threadIdx.x, wave = tid >> 5, lane = tid & 31;
    int hi = lane >> 4, lm = lane & 15;
    for (int idx = tid; idx < DIM*DIM; idx += 256) sW[idx] = Wp[idx];
    __syncthreads();

    int b    = blockIdx.x >> 11;
    int pix0 = (blockIdx.x & 2047) * 32;
    int mt = wave & 3, nt = wave >> 2;

    f32x8 acc;
    #pragma unroll
    for (int v = 0; v < 8; ++v) acc[v] = b_proj[mt*16 + v + 8*hi];

    #pragma unroll
    for (int kk = 0; kk < DIM; kk += 32) {
        frag_t a, bf;
        a.h[0] = *(const bf16x8*)&sW[(mt*16 + lm)*DIM + kk + hi*8];
        a.h[1] = *(const bf16x8*)&sW[(mt*16 + lm)*DIM + kk + 16 + hi*8];
        #pragma unroll
        for (int e = 0; e < 16; ++e) bf.v[e] = in[(size_t)(b*DIM + kk + hi*8 + e + (e & 8))*NPIX + pix0 + nt*16 + lm];
        acc = wmma_bf16(a.v, bf.v, acc);
    }
    #pragma unroll
    for (int v = 0; v < 8; ++v) {
        int co = mt*16 + v + 8*hi;
        out[(size_t)(b*DIM + co)*NPIX + pix0 + nt*16 + lm] = acc[v];
    }
}

extern "C" void kernel_launch(void* const* d_in, const int* in_sizes, int n_in,
                              void* d_out, int out_size, void* d_ws, size_t ws_size,
                              hipStream_t stream) {
    (void)in_sizes; (void)n_in; (void)out_size; (void)ws_size;
    const float* x           = (const float*)d_in[0];
    const float* temperature = (const float*)d_in[1];
    const float* w_qkv       = (const float*)d_in[2];
    const float* b_qkv       = (const float*)d_in[3];
    const float* w_q         = (const float*)d_in[4];
    const float* b_q         = (const float*)d_in[5];
    const float* w_k         = (const float*)d_in[6];
    const float* b_k         = (const float*)d_in[7];
    const float* w_dwv       = (const float*)d_in[8];
    const float* b_dwv       = (const float*)d_in[9];
    const float* w_proj      = (const float*)d_in[10];
    const float* b_proj      = (const float*)d_in[11];
    float* out = (float*)d_out;
    char*  ws  = (char*)d_ws;

    fold_weights<<<1, 256, 0, stream>>>(w_qkv, b_qkv, w_q, b_q, w_k, b_k, w_proj, ws);
    qk_kernel<<<4096, 256, 0, stream>>>(x, temperature, ws);
    v0_wmma<<<BATCH*2048, 256, 0, stream>>>(x, ws);
    dw3x3<<<65536, 256, 0, stream>>>(w_dwv, b_dwv, ws);
    attn_wmma<<<BATCH*DIM, 256, 0, stream>>>(ws);
    proj_wmma<<<BATCH*2048, 256, 0, stream>>>(b_proj, out, ws);
}
